// GraphAttentionLayer_71519795413773
// MI455X (gfx1250) — matrix operand order repaired and checked by reading, not compiled, not run on hardware
//
#include <hip/hip_runtime.h>

typedef __bf16 v8bf  __attribute__((ext_vector_type(8)));
typedef __bf16 v16bf __attribute__((ext_vector_type(16)));
typedef float  v8f   __attribute__((ext_vector_type(8)));

namespace {
constexpr int   kDM    = 512;
constexpr int   kSROWH = 520;   // bf16-plane row stride (elements)
constexpr int   kSROWF = 520;   // fp32 LDS row stride (floats)
constexpr int   kMT    = 32;    // tokens per workgroup
constexpr float kEps   = 1e-5f;
constexpr float kScale = 0.088388347648318447f;  // 1/sqrt(128)
constexpr int   kWElem = kDM * kDM;              // 262144 elements per weight
}

union V16U {
  v16bf v;
  v8bf  h[2];
};

__device__ __forceinline__ v8f wmma_bf16(v16bf a, v16bf b, v8f c) {
  // V_WMMA_F32_16X16X32_BF16 : D = A(16x32,bf16) * B(32x16,bf16) + C(16x16,f32)
  return __builtin_amdgcn_wmma_f32_16x16x32_bf16(false, a, false, b, (short)0, c,
                                                 false, false);
}

// ---------------------------------------------------------------------------
// Weight split: W(fp32) -> W_hi(bf16) + W_lo(bf16 of residual)
// ---------------------------------------------------------------------------
__global__ void split_weights_kernel(const float* __restrict__ W,
                                     __bf16* __restrict__ Wh,
                                     __bf16* __restrict__ Wl, int n) {
  const int i = blockIdx.x * 256 + threadIdx.x;
  if (i < n) {
    const float w = W[i];
    const __bf16 h = (__bf16)w;
    Wh[i] = h;
    Wl[i] = (__bf16)(w - (float)h);
  }
}

// ---------------------------------------------------------------------------
// D[32x512] = A @ W^T + bias via split-bf16 (3 WMMAs per 32-wide K chunk).
// Wave `wid` owns columns {wid*16 + ct*128 + 0..15 | ct = 0..3}.
// A layout: lane(m,half) comps 0..15 = K[k0+16h .. +15].
// B: lane(n,half) comps = K[k0+16h .. +15].
// ---------------------------------------------------------------------------
__device__ __forceinline__ void gemm_bf16x3(const __bf16* __restrict__ Wh,
                                            const __bf16* __restrict__ Wl,
                                            const float* __restrict__ Wb,
                                            const __bf16* Ah, const __bf16* Al,
                                            float* D, int wid, int lane) {
  const int half = lane >> 4;
  const int l16  = lane & 15;
  v8f acc[4][2];
#pragma unroll
  for (int ct = 0; ct < 4; ++ct) {
    acc[ct][0] = {};
    acc[ct][1] = {};
  }
  const int rowA0 = l16 * kSROWH;
  const int rowA1 = (l16 + 16) * kSROWH;
  for (int k0 = 0; k0 < kDM; k0 += 32) {
    const int ka = k0 + 16 * half;
    V16U ah0, ah1, al0, al1;
    ah0.h[0] = *(const v8bf*)(Ah + rowA0 + ka);
    ah0.h[1] = *(const v8bf*)(Ah + rowA0 + ka + 8);
    ah1.h[0] = *(const v8bf*)(Ah + rowA1 + ka);
    ah1.h[1] = *(const v8bf*)(Ah + rowA1 + ka + 8);
    al0.h[0] = *(const v8bf*)(Al + rowA0 + ka);
    al0.h[1] = *(const v8bf*)(Al + rowA0 + ka + 8);
    al1.h[0] = *(const v8bf*)(Al + rowA1 + ka);
    al1.h[1] = *(const v8bf*)(Al + rowA1 + ka + 8);
    const int kb = k0 + 16 * half;
#pragma unroll
    for (int ct = 0; ct < 4; ++ct) {
      const size_t n = (size_t)(wid * 16 + ct * 128 + l16);
      const v16bf bh = *(const v16bf*)(Wh + n * kDM + kb);
      const v16bf bl = *(const v16bf*)(Wl + n * kDM + kb);
      // rows 0..15
      acc[ct][0] = wmma_bf16(ah0.v, bh, acc[ct][0]);
      acc[ct][0] = wmma_bf16(ah0.v, bl, acc[ct][0]);
      acc[ct][0] = wmma_bf16(al0.v, bh, acc[ct][0]);
      // rows 16..31
      acc[ct][1] = wmma_bf16(ah1.v, bh, acc[ct][1]);
      acc[ct][1] = wmma_bf16(ah1.v, bl, acc[ct][1]);
      acc[ct][1] = wmma_bf16(al1.v, bh, acc[ct][1]);
    }
  }
#pragma unroll
  for (int ct = 0; ct < 4; ++ct) {
    const int n = wid * 16 + ct * 128 + l16;
    const float bias = Wb[n];
#pragma unroll
    for (int v = 0; v < 8; ++v) {
      const int m = v + 8 * half;  // D: lane = N, VGPR v = M (+8 per half)
      D[m * kSROWF + n]        = acc[ct][0][v] + bias;
      D[(m + 16) * kSROWF + n] = acc[ct][1][v] + bias;
    }
  }
}

__global__ __launch_bounds__(256, 1) void fused_attn_ln_kernel(
    const float* __restrict__ x,
    const __bf16* __restrict__ Wsplit,  // [Wq,Wk,Wv,Wo] x [hi,lo] planes
    const float* __restrict__ Wq_b, const float* __restrict__ Wk_b,
    const float* __restrict__ Wv_b, const float* __restrict__ Wo_b,
    const float* __restrict__ ln_g, const float* __restrict__ ln_b,
    float* __restrict__ out) {
  extern __shared__ unsigned char smem_raw[];
  __bf16* xh = (__bf16*)smem_raw;            // [32][520] x hi plane
  __bf16* xl = xh + kMT * kSROWH;            // [32][520] x lo plane
  float*  qs = (float*)(xl + kMT * kSROWH);  // Q fp32, later attn_out hi/lo
  float*  ks = qs + kMT * kSROWF;            // K fp32, later y (pre-LN)
  float*  vs = ks + kMT * kSROWF;            // V fp32
  float*  sc = vs + kMT * kSROWF;            // [32][16] scores -> attn

  const __bf16* Wq_h = Wsplit + 0 * kWElem;
  const __bf16* Wq_l = Wsplit + 1 * kWElem;
  const __bf16* Wk_h = Wsplit + 2 * kWElem;
  const __bf16* Wk_l = Wsplit + 3 * kWElem;
  const __bf16* Wv_h = Wsplit + 4 * kWElem;
  const __bf16* Wv_l = Wsplit + 5 * kWElem;
  const __bf16* Wo_h = Wsplit + 6 * kWElem;
  const __bf16* Wo_l = Wsplit + 7 * kWElem;

  const int tid  = threadIdx.x;
  const int wid  = tid >> 5;
  const int lane = tid & 31;
  const long tok0 = (long)blockIdx.x * kMT;

  // ---- stage x tile, split to hi/lo bf16 planes ----
#pragma unroll
  for (int i = 0; i < 16; ++i) {
    const int idx = tid + i * 256;  // 0..4095
    const int r = idx >> 7, c4 = idx & 127;
    const float4 v = *(const float4*)(x + (tok0 + r) * kDM + c4 * 4);
    const int o = r * kSROWH + c4 * 4;
    const __bf16 h0 = (__bf16)v.x, h1 = (__bf16)v.y;
    const __bf16 h2 = (__bf16)v.z, h3 = (__bf16)v.w;
    xh[o + 0] = h0; xh[o + 1] = h1; xh[o + 2] = h2; xh[o + 3] = h3;
    xl[o + 0] = (__bf16)(v.x - (float)h0);
    xl[o + 1] = (__bf16)(v.y - (float)h1);
    xl[o + 2] = (__bf16)(v.z - (float)h2);
    xl[o + 3] = (__bf16)(v.w - (float)h3);
  }
  __syncthreads();

  // ---- Q, K, V projections (split-bf16 WMMA) ----
  gemm_bf16x3(Wq_h, Wq_l, Wq_b, xh, xl, qs, wid, lane);
  gemm_bf16x3(Wk_h, Wk_l, Wk_b, xh, xl, ks, wid, lane);
  gemm_bf16x3(Wv_h, Wv_l, Wv_b, xh, xl, vs, wid, lane);
  __syncthreads();

  // ---- per-token head-axis scores: sc[r][h*4+t] = Q[r,h,:]·K[r,t,:]/sqrt(128)
#pragma unroll
  for (int i = 0; i < 2; ++i) {
    const int idx = tid + i * 256;  // 0..511
    const int r = idx >> 4, p = idx & 15;
    const int h = p >> 2, t = p & 3;
    const float* qrow = qs + r * kSROWF + h * 128;
    const float* krow = ks + r * kSROWF + t * 128;
    float acc = 0.f;
#pragma unroll 4
    for (int d = 0; d < 128; ++d) acc = fmaf(qrow[d], krow[d], acc);
    sc[r * 16 + p] = acc * kScale;
  }
  __syncthreads();

  // ---- softmax over the 4-wide head axis ----
  if (tid < kMT * 4) {
    const int r = tid >> 2, h = tid & 3;
    float* s = sc + r * 16 + h * 4;
    const float s0 = s[0], s1 = s[1], s2 = s[2], s3 = s[3];
    const float m = fmaxf(fmaxf(s0, s1), fmaxf(s2, s3));
    const float e0 = __expf(s0 - m), e1 = __expf(s1 - m);
    const float e2 = __expf(s2 - m), e3 = __expf(s3 - m);
    const float inv = 1.f / (e0 + e1 + e2 + e3);
    s[0] = e0 * inv; s[1] = e1 * inv; s[2] = e2 * inv; s[3] = e3 * inv;
  }
  __syncthreads();

  // ---- attn_out = attn @ V, re-split to hi/lo planes (overwrite Q region) ----
  {
    __bf16* aoh = (__bf16*)qs;              // [32][520] hi plane
    __bf16* aol = aoh + kMT * kSROWH;       // [32][520] lo plane
    const int r = tid >> 3, s8 = tid & 7;
    const float* vrow = vs + r * kSROWF;
#pragma unroll 4
    for (int j = 0; j < 64; ++j) {
      const int c = s8 + 8 * j;
      const int h = c >> 7, d = c & 127;
      const float* a = sc + r * 16 + h * 4;
      const float val = a[0] * vrow[d]       + a[1] * vrow[128 + d] +
                        a[2] * vrow[256 + d] + a[3] * vrow[384 + d];
      const __bf16 hi = (__bf16)val;
      aoh[r * kSROWH + c] = hi;
      aol[r * kSROWH + c] = (__bf16)(val - (float)hi);
    }
  }
  __syncthreads();

  // ---- output projection: y = attn_out @ Wo^T + bo  (into K region) ----
  {
    const __bf16* aoh = (const __bf16*)qs;
    const __bf16* aol = aoh + kMT * kSROWH;
    gemm_bf16x3(Wo_h, Wo_l, Wo_b, aoh, aol, ks, wid, lane);
  }
  __syncthreads();

  // ---- residual (x = xh + xl) + LayerNorm, 8 lanes/row, width-8 shuffles ----
  {
    const int r = tid >> 3, s8 = tid & 7;
    const float* yrow = ks + r * kSROWF;
    const __bf16* xhr = xh + r * kSROWH;
    const __bf16* xlr = xl + r * kSROWH;
    float sum = 0.f, sq = 0.f;
#pragma unroll 8
    for (int j = 0; j < 64; ++j) {
      const int c = s8 + 8 * j;
      const float y = yrow[c] + (float)xhr[c] + (float)xlr[c];
      sum += y;
      sq = fmaf(y, y, sq);
    }
#pragma unroll
    for (int off = 1; off < 8; off <<= 1) {
      sum += __shfl_xor(sum, off, 8);
      sq  += __shfl_xor(sq, off, 8);
    }
    const float mean = sum * (1.f / kDM);
    const float var  = fmaf(mean, -mean, sq * (1.f / kDM));
    const float rstd = rsqrtf(var + kEps);
    float* orow = out + (tok0 + r) * kDM;
#pragma unroll 4
    for (int j = 0; j < 64; ++j) {
      const int c = s8 + 8 * j;
      const float y = yrow[c] + (float)xhr[c] + (float)xlr[c];
      orow[c] = (y - mean) * rstd * ln_g[c] + ln_b[c];
    }
  }
}

extern "C" void kernel_launch(void* const* d_in, const int* in_sizes, int n_in,
                              void* d_out, int out_size, void* d_ws, size_t ws_size,
                              hipStream_t stream) {
  const float* x    = (const float*)d_in[0];
  const float* Wq_w = (const float*)d_in[1];
  const float* Wq_b = (const float*)d_in[2];
  const float* Wk_w = (const float*)d_in[3];
  const float* Wk_b = (const float*)d_in[4];
  const float* Wv_w = (const float*)d_in[5];
  const float* Wv_b = (const float*)d_in[6];
  const float* Wo_w = (const float*)d_in[7];
  const float* Wo_b = (const float*)d_in[8];
  const float* ln_g = (const float*)d_in[9];
  const float* ln_b = (const float*)d_in[10];
  float* out = (float*)d_out;

  // Workspace: [Wq,Wk,Wv,Wo] x [hi,lo] bf16 planes, 4 MB total.
  __bf16* Wsplit = (__bf16*)d_ws;
  const float* Wsrc[4] = {Wq_w, Wk_w, Wv_w, Wo_w};
  const dim3 cgrid((kWElem + 255) / 256), cblock(256);
  for (int m = 0; m < 4; ++m) {
    split_weights_kernel<<<cgrid, cblock, 0, stream>>>(
        Wsrc[m], Wsplit + (2 * m) * kWElem, Wsplit + (2 * m + 1) * kWElem,
        kWElem);
  }

  const int ntok = in_sizes[0] / kDM;  // B*S = 65536
  const size_t shmem = (size_t)(2 * kMT * kSROWH) * sizeof(__bf16) +
                       (size_t)(3 * kMT * kSROWF + kMT * 16) * sizeof(float);
  dim3 grid(ntok / kMT), block(256);
  fused_attn_ln_kernel<<<grid, block, shmem, stream>>>(
      x, Wsplit, Wq_b, Wk_b, Wv_b, Wo_b, ln_g, ln_b, out);
}
